// KPConv_76424648066043
// MI455X (gfx1250) — matrix operand order repaired and checked by reading, not compiled, not run on hardware
//
#include <hip/hip_runtime.h>
#include <hip/hip_bf16.h>
#include <math.h>

// ---------------- problem constants (from reference) ----------------
#define NQ     25000   // queries
#define MPTS   25000   // support points
#define HN     32      // neighbors
#define KP     15      // kernel points (padded to 16 for WMMA)
#define CIN    128
#define COUT   128
#define KP_EXT 1.2f

typedef _Float16 h8  __attribute__((ext_vector_type(8)));
typedef _Float16 h16 __attribute__((ext_vector_type(16)));
typedef float    f8  __attribute__((ext_vector_type(8)));

static __device__ __forceinline__ h16 cat16(h8 lo, h8 hi) {
    h16 r;
#pragma unroll
    for (int i = 0; i < 8; ++i) { r[i] = lo[i]; r[i + 8] = hi[i]; }
    return r;
}

// --------------------------------------------------------------------
// Pack weights [15,128,128] f32 -> padded f16 WMMA-B fragments.
// R = k*128 + c (k padded to 16 -> R in [0,2048)), output cols d.
// Chunk (r, nb): 32 (K-dim) x 16 (N-dim) tile; lane L holds column
// n = L%16, K-rows 8*(L/16)+(s&7)+16*(s>>3) (s=0..15) stored per lane, so each
// lane's 16 halves are 32 contiguous bytes -> 2x global_load_b128.
// Total: 512 chunks * 512 halves = 512 KB (L2-resident).
// --------------------------------------------------------------------
__global__ void kpconv_pack_weights(const float* __restrict__ W,
                                    _Float16* __restrict__ Wp) {
    int o = blockIdx.x * 256 + threadIdx.x;          // < 262144
    int chunk  = o >> 9;                             // /512
    int within = o & 511;
    int lane = within >> 4;
    int s    = within & 15;
    int r  = chunk >> 3;
    int nb = chunk & 7;
    int n  = lane & 15;
    int hh = lane >> 4;
    int R  = 32 * r + 8 * hh + (s & 7) + 16 * (s >> 3); // flattened (k,c)
    int k  = R >> 7;
    int c  = R & 127;
    int d  = 16 * nb + n;
    float v = (k < KP) ? W[((size_t)k * CIN + c) * COUT + d] : 0.0f;
    Wp[o] = (_Float16)v;
}

// --------------------------------------------------------------------
// Main fused KPConv kernel. One workgroup (8 waves) per 16 queries.
// --------------------------------------------------------------------
__global__ void __launch_bounds__(256)
kpconv_kernel(const float* __restrict__ x,
              const float* __restrict__ q_pts,
              const float* __restrict__ s_pts,
              const long long* __restrict__ inds,
              const float* __restrict__ kpts,
              const _Float16* __restrict__ Wp,
              float* __restrict__ out) {
    __shared__ __align__(16) float    s_qpt[16][3];
    __shared__ __align__(16) int      s_idx[16][HN];           //  2 KB (clamped ids)
    __shared__ __align__(16) float    s_rel[16][HN][3];        //  6 KB
    __shared__ __align__(16) _Float16 s_w[16][16][HN];         // 16 KB  w[q][k][h]
    __shared__ __align__(16) _Float16 s_wf[16][16][CIN];       // 64 KB  wf[q][k][c]

    const int tid  = threadIdx.x;
    const int lane = tid & 31;
    const int wv   = tid >> 5;        // wave id 0..7
    const int lm   = lane & 15;       // row/col within 16
    const int lh   = lane >> 4;       // lane half
    const int n0   = blockIdx.x * 16; // first query of tile

    // ---- Phase 0a: query points ----
    if (tid < 16) {
        int qg = n0 + tid;
        float px = 0.f, py = 0.f, pz = 0.f;
        if (qg < NQ) {
            px = q_pts[qg * 3 + 0];
            py = q_pts[qg * 3 + 1];
            pz = q_pts[qg * 3 + 2];
        }
        s_qpt[tid][0] = px; s_qpt[tid][1] = py; s_qpt[tid][2] = pz;
    }
    __syncthreads();

    // ---- Phase 0b: neighbor indices + relative positions ----
    // Shadow neighbors (id==M) get position 1e6 -> influence weight is
    // exactly 0 for every kernel point, so the feature row they gather is
    // multiplied by an all-zero WMMA A-column. Hence we may CLAMP the index
    // for the feature gather (no branch needed later in the hot loop).
    for (int e = tid; e < 16 * HN; e += 256) {
        int q = e >> 5, h = e & 31;
        int qg = n0 + q;
        int id = MPTS; // shadow for out-of-range queries
        if (qg < NQ) id = (int)inds[(size_t)qg * HN + h];
        float px, py, pz;
        if (id < MPTS) {
            px = s_pts[id * 3 + 0]; py = s_pts[id * 3 + 1]; pz = s_pts[id * 3 + 2];
        } else {
            px = py = pz = 1.0e6f;  // shadow point far away -> weight clips to 0
        }
        s_rel[q][h][0] = px - s_qpt[q][0];
        s_rel[q][h][1] = py - s_qpt[q][1];
        s_rel[q][h][2] = pz - s_qpt[q][2];
        s_idx[q][h] = (id < MPTS) ? id : (MPTS - 1);   // clamped for gather
    }
    __syncthreads();

    // ---- Phase 0c: influence weights w[q][k][h] (k=15 zero pad) ----
    for (int e = tid; e < 16 * 16 * HN; e += 256) {
        int q = e >> 9;
        int k = (e >> 5) & 15;
        int h = e & 31;
        _Float16 wq = (_Float16)0.0f;
        if (k < KP) {
            float dx = s_rel[q][h][0] - kpts[k * 3 + 0];
            float dy = s_rel[q][h][1] - kpts[k * 3 + 1];
            float dz = s_rel[q][h][2] - kpts[k * 3 + 2];
            float d2 = dx * dx + dy * dy + dz * dz;
            float wf = 1.0f - sqrtf(d2) * (1.0f / KP_EXT);
            wq = (_Float16)fmaxf(wf, 0.0f);
        }
        s_w[q][k][h] = wq;
    }
    __syncthreads();

    // ---- Phase B: per-query [16k x 32h] x [32h x 128c] via 8 WMMAs ----
    // A layout (16-bit 16x32): lane m=L%16; slots 0..7 = h in
    // [8*half, 8*half+8), slots 8..15 = +16.
    // B layout: lane n=L%16; slots 0..7 = K-rows 8*half+s, slots 8..15 = +16.
#pragma unroll
    for (int round = 0; round < 2; ++round) {
        const int q = wv + 8 * round;
        const h8* wrow = (const h8*)&s_w[q][lm][0];   // 4 groups of 8 halves
        h16 afrag = cat16(wrow[lh], wrow[2 + lh]);

        // Hoist the 16 gather-row element offsets out of the block loop:
        // unconditional (clamped) -> compiler batches independent loads.
        unsigned offs[16];
#pragma unroll
        for (int s = 0; s < 16; ++s)
            offs[s] = (unsigned)s_idx[q][8 * lh + (s & 7) + 16 * (s >> 3)] * CIN;

#pragma unroll
        for (int b = 0; b < 8; ++b) {
            const int c = 16 * b + lm;
            h16 bfrag;
#pragma unroll
            for (int s = 0; s < 16; ++s)
                bfrag[s] = (_Float16)x[offs[s] + c];

            f8 acc = {};
            acc = __builtin_amdgcn_wmma_f32_16x16x32_f16(
                false, afrag, false, bfrag, (short)0, acc, false, false);
            // D layout: lane col n=L%16 (=c), row m=j+8*half (=k)
#pragma unroll
            for (int j = 0; j < 8; ++j) {
                int k = j + 8 * lh;
                s_wf[q][k][c] = (_Float16)acc[j];
            }
        }
    }
    __syncthreads();

    // ---- Phase C: [16q x 2048R] x [2048R x 128d], wave wv owns d-block ----
    f8 acc = {};
#pragma unroll 4
    for (int r = 0; r < 64; ++r) {
        const int k  = r >> 2;
        const int g0 = (r & 3) * 4;                   // c0/8 in h8 units
        const h8* row = (const h8*)&s_wf[lm][k][0];   // 16 h8 groups
        h16 afrag = cat16(row[g0 + lh], row[g0 + 2 + lh]);
        const h8* wp = (const h8*)(Wp + (((size_t)(r * 8 + wv)) << 9) + lane * 16);
        h16 bfrag = cat16(wp[0], wp[1]);
        acc = __builtin_amdgcn_wmma_f32_16x16x32_f16(
            false, afrag, false, bfrag, (short)0, acc, false, false);
    }
    // D: lane col n = d within block, row m = query j+8*half
#pragma unroll
    for (int j = 0; j < 8; ++j) {
        int m  = j + 8 * lh;
        int qg = n0 + m;
        if (qg < NQ) out[(size_t)qg * COUT + 16 * wv + lm] = acc[j];
    }
}

// --------------------------------------------------------------------
extern "C" void kernel_launch(void* const* d_in, const int* in_sizes, int n_in,
                              void* d_out, int out_size, void* d_ws, size_t ws_size,
                              hipStream_t stream) {
    (void)in_sizes; (void)n_in; (void)out_size; (void)ws_size;
    const float*     x    = (const float*)d_in[0];
    const float*     qp   = (const float*)d_in[1];
    const float*     sp   = (const float*)d_in[2];
    const long long* inds = (const long long*)d_in[3];
    const float*     kpts = (const float*)d_in[4];
    const float*     W    = (const float*)d_in[5];
    float*           out  = (float*)d_out;
    _Float16*        Wp   = (_Float16*)d_ws;          // 512 KB packed weights

    // One-time (per launch, deterministic) weight repack: 262144 elements.
    kpconv_pack_weights<<<1024, 256, 0, stream>>>(W, Wp);

    const int nblocks = (NQ + 15) / 16;               // 1563
    kpconv_kernel<<<nblocks, 256, 0, stream>>>(x, qp, sp, inds, kpts, Wp, out);
}
